// MLPEncoder_91061896610585
// MI455X (gfx1250) — matrix operand order repaired and checked by reading, not compiled, not run on hardware
//
#include <hip/hip_runtime.h>
#include <hip/hip_bf16.h>

// ---------------------------------------------------------------------------
// Fused MLP (128->256->256->128, ReLU) with obs-select, bf16 WMMA, two passes
// (pos / neg). Each wave owns 64 rows (4 M-tiles) so every 1KB B-fragment
// feeds 4 v_wmma_f32_16x16x32_bf16 issues (4x less L2 weight traffic).
// Weights pre-packed into WMMA B-fragment layout in d_ws (bf16, 512KB).
// ---------------------------------------------------------------------------

typedef __bf16 bf16_t;
typedef __attribute__((ext_vector_type(16))) __bf16 v16bf;
typedef __attribute__((ext_vector_type(8)))  __bf16 v8bf;
typedef __attribute__((ext_vector_type(8)))  float  v8f;

#define AROW 264            // LDS activation row stride (bf16 elems), 16B aligned, staggers banks

// ws fragment-region offsets, in bf16 elements (each fragment = 512 bf16 = 1KB)
#define OFF_PW0 0           //  4 kt x 16 nt
#define OFF_PW1 32768       //  8 kt x 16 nt
#define OFF_PW2 98304       //  8 kt x  8 nt
#define OFF_NW0 131072
#define OFF_NW1 163840
#define OFF_NW2 229376
#define WS_ELEMS 262144     // 512 KB of bf16

#define T_DIM 8
#define N_DIM 50000
#define D_DIM 128
#define ROWS_PER_WAVE 64    // 4 M-tiles of 16

// ---------------------------------------------------------------------------
// Pack fp32 weights -> bf16 WMMA B-fragments.
// B layout (32x16, K x N): lane l holds column N = l%16; lanes 0-15 hold
// K {0..7,16..23}, lanes 16-31 hold K {8..15,24..31} (matches A layout).
// Fragment storage: elem index = lane*16 + slot (32B contiguous per lane).
// Fragment order within a matrix: frag = kt*NT + nt.
// ---------------------------------------------------------------------------
__global__ __launch_bounds__(256) void pack_weights(
    const float* __restrict__ pw0, const float* __restrict__ pw1, const float* __restrict__ pw2,
    const float* __restrict__ nw0, const float* __restrict__ nw1, const float* __restrict__ nw2,
    bf16_t* __restrict__ ws)
{
    int i = blockIdx.x * blockDim.x + threadIdx.x;
    if (i >= WS_ELEMS) return;

    const float* W; int NT; int off; int r = i;
    if      (r < OFF_PW1) { W = pw0; NT = 16; off = OFF_PW0; r -= OFF_PW0; }
    else if (r < OFF_PW2) { W = pw1; NT = 16; off = OFF_PW1; r -= OFF_PW1; }
    else if (r < OFF_NW0) { W = pw2; NT =  8; off = OFF_PW2; r -= OFF_PW2; }
    else if (r < OFF_NW1) { W = nw0; NT = 16; off = OFF_NW0; r -= OFF_NW0; }
    else if (r < OFF_NW2) { W = nw1; NT = 16; off = OFF_NW1; r -= OFF_NW1; }
    else                  { W = nw2; NT =  8; off = OFF_NW2; r -= OFF_NW2; }

    const int frag   = r >> 9;          // /512
    const int within = r & 511;
    const int lane   = within >> 4;
    const int s      = within & 15;
    const int kt     = frag / NT;
    const int nt     = frag - kt * NT;

    const int k    = kt * 32 + (s & 7) + ((s >> 3) << 4) + ((lane >= 16) ? 8 : 0);
    const int n    = nt * 16 + (lane & 15);
    const int dout = NT * 16;           // 256 or 128; W is row-major [din][dout]

    ws[off + frag * 512 + within] = (bf16_t)W[k * dout + n];
}

// ---------------------------------------------------------------------------
// One MLP pass over 64 rows per wave (one wave per block; no barriers needed,
// per-wave DS ordering covers the LDS restripe between layers).
//   sel = +1: write MLP result where obs==sel; if writeZero, also copy
//             node_data where obs==0.  sel = -1: neg pass.
// ---------------------------------------------------------------------------
__global__ __launch_bounds__(32) void mlp_pass(
    const float* __restrict__ node, const int* __restrict__ obs,
    const bf16_t* __restrict__ W0, const bf16_t* __restrict__ W1, const bf16_t* __restrict__ W2,
    const float* __restrict__ b0, const float* __restrict__ b1, const float* __restrict__ b2,
    float* __restrict__ out, int sel, int writeZero)
{
    __shared__ __align__(16) bf16_t act[ROWS_PER_WAVE * AROW];   // 64 x 256 bf16 (padded)

    const int lane = threadIdx.x & 31;
    const int m    = lane & 15;          // row (A layout) or column (B/C layout)
    const int hi   = lane >> 4;          // 0 = lanes 0-15, 1 = lanes 16-31
    const int row0 = blockIdx.x * ROWS_PER_WAVE;

    // A fragments: [kt][mt], kept in registers across each layer's nt loop.
    v16bf A[8][4];

    // ---- Layer-0 A fragments from node_data (fp32 -> bf16).
    // A layout: lane holds row m; lanes 0-15 slots = K {0..7,16..23},
    // lanes 16-31 slots = K {8..15,24..31}.
    #pragma unroll
    for (int mt = 0; mt < 4; ++mt) {
        const float* rowp = node + (size_t)(row0 + mt * 16 + m) * D_DIM;
        #pragma unroll
        for (int kt = 0; kt < 4; ++kt) {
            const int k1 = kt * 32 + hi * 8;
            const int k2 = kt * 32 + 16 + hi * 8;
            float4 f0 = *(const float4*)(rowp + k1);
            float4 f1 = *(const float4*)(rowp + k1 + 4);
            float4 f2 = *(const float4*)(rowp + k2);
            float4 f3 = *(const float4*)(rowp + k2 + 4);
            v16bf a;
            a[0]=(bf16_t)f0.x;  a[1]=(bf16_t)f0.y;  a[2]=(bf16_t)f0.z;  a[3]=(bf16_t)f0.w;
            a[4]=(bf16_t)f1.x;  a[5]=(bf16_t)f1.y;  a[6]=(bf16_t)f1.z;  a[7]=(bf16_t)f1.w;
            a[8]=(bf16_t)f2.x;  a[9]=(bf16_t)f2.y;  a[10]=(bf16_t)f2.z; a[11]=(bf16_t)f2.w;
            a[12]=(bf16_t)f3.x; a[13]=(bf16_t)f3.y; a[14]=(bf16_t)f3.z; a[15]=(bf16_t)f3.w;
            A[kt][mt] = a;
        }
    }

    // Re-stripe LDS activations (D layout) -> A fragments for next layer.
    auto restripe = [&]() {
        #pragma unroll
        for (int kt = 0; kt < 8; ++kt) {
            #pragma unroll
            for (int mt = 0; mt < 4; ++mt) {
                const bf16_t* p = act + (mt * 16 + m) * AROW;
                v8bf lo = *(const v8bf*)(p + kt * 32 + hi * 8);
                v8bf hh = *(const v8bf*)(p + kt * 32 + 16 + hi * 8);
                A[kt][mt] = __builtin_shufflevector(lo, hh,
                    0,1,2,3,4,5,6,7,8,9,10,11,12,13,14,15);
            }
        }
    };

    // ================= Layer 0: 128 -> 256, ReLU =================
    #pragma unroll 1
    for (int nt = 0; nt < 16; ++nt) {
        const float bb = b0[nt * 16 + m];
        v8f c[4];
        #pragma unroll
        for (int mt = 0; mt < 4; ++mt) { v8f ci = {bb,bb,bb,bb,bb,bb,bb,bb}; c[mt] = ci; }
        #pragma unroll
        for (int kt = 0; kt < 4; ++kt) {
            v16bf B = *(const v16bf*)(W0 + (((kt * 16 + nt) << 9) + (lane << 4)));
            #pragma unroll
            for (int mt = 0; mt < 4; ++mt)
                c[mt] = __builtin_amdgcn_wmma_f32_16x16x32_bf16(
                            false, A[kt][mt], false, B, (short)0, c[mt], false, false);
        }
        #pragma unroll
        for (int mt = 0; mt < 4; ++mt)
            #pragma unroll
            for (int r = 0; r < 8; ++r)
                act[(mt * 16 + r + hi * 8) * AROW + nt * 16 + m] = (bf16_t)fmaxf(c[mt][r], 0.0f);
    }

    restripe();

    // ================= Layer 1: 256 -> 256, ReLU =================
    #pragma unroll 1
    for (int nt = 0; nt < 16; ++nt) {
        const float bb = b1[nt * 16 + m];
        v8f c[4];
        #pragma unroll
        for (int mt = 0; mt < 4; ++mt) { v8f ci = {bb,bb,bb,bb,bb,bb,bb,bb}; c[mt] = ci; }
        #pragma unroll
        for (int kt = 0; kt < 8; ++kt) {
            v16bf B = *(const v16bf*)(W1 + (((kt * 16 + nt) << 9) + (lane << 4)));
            #pragma unroll
            for (int mt = 0; mt < 4; ++mt)
                c[mt] = __builtin_amdgcn_wmma_f32_16x16x32_bf16(
                            false, A[kt][mt], false, B, (short)0, c[mt], false, false);
        }
        #pragma unroll
        for (int mt = 0; mt < 4; ++mt)
            #pragma unroll
            for (int r = 0; r < 8; ++r)
                act[(mt * 16 + r + hi * 8) * AROW + nt * 16 + m] = (bf16_t)fmaxf(c[mt][r], 0.0f);
    }

    restripe();

    // ---- observations: 8 consecutive rows per (mt, half-wave) -> int4 loads
    int ov[4][8];
    #pragma unroll
    for (int mt = 0; mt < 4; ++mt) {
        const int* op = obs + row0 + mt * 16 + hi * 8;
        int4 o0 = *(const int4*)(op);
        int4 o1 = *(const int4*)(op + 4);
        ov[mt][0] = o0.x; ov[mt][1] = o0.y; ov[mt][2] = o0.z; ov[mt][3] = o0.w;
        ov[mt][4] = o1.x; ov[mt][5] = o1.y; ov[mt][6] = o1.z; ov[mt][7] = o1.w;
    }

    // ================= Layer 2: 256 -> 128, linear + select + store =================
    #pragma unroll 1
    for (int nt = 0; nt < 8; ++nt) {
        const float bb = b2[nt * 16 + m];
        v8f c[4];
        #pragma unroll
        for (int mt = 0; mt < 4; ++mt) { v8f ci = {bb,bb,bb,bb,bb,bb,bb,bb}; c[mt] = ci; }
        #pragma unroll
        for (int kt = 0; kt < 8; ++kt) {
            v16bf B = *(const v16bf*)(W2 + (((kt * 8 + nt) << 9) + (lane << 4)));
            #pragma unroll
            for (int mt = 0; mt < 4; ++mt)
                c[mt] = __builtin_amdgcn_wmma_f32_16x16x32_bf16(
                            false, A[kt][mt], false, B, (short)0, c[mt], false, false);
        }
        #pragma unroll
        for (int mt = 0; mt < 4; ++mt) {
            #pragma unroll
            for (int r = 0; r < 8; ++r) {
                const int o   = ov[mt][r];
                const int row = row0 + mt * 16 + r + hi * 8;
                const int col = nt * 16 + m;
                if (o == sel) {
                    out[(size_t)row * D_DIM + col] = c[mt][r];
                } else if (writeZero && o == 0) {
                    out[(size_t)row * D_DIM + col] = node[(size_t)row * D_DIM + col];
                }
            }
        }
    }
}

// ---------------------------------------------------------------------------
extern "C" void kernel_launch(void* const* d_in, const int* in_sizes, int n_in,
                              void* d_out, int out_size, void* d_ws, size_t ws_size,
                              hipStream_t stream) {
    const float* node = (const float*)d_in[0];
    const int*   obs  = (const int*)d_in[1];
    const float* pw0  = (const float*)d_in[2];  const float* pb0 = (const float*)d_in[3];
    const float* pw1  = (const float*)d_in[4];  const float* pb1 = (const float*)d_in[5];
    const float* pw2  = (const float*)d_in[6];  const float* pb2 = (const float*)d_in[7];
    const float* nw0  = (const float*)d_in[8];  const float* nb0 = (const float*)d_in[9];
    const float* nw1  = (const float*)d_in[10]; const float* nb1 = (const float*)d_in[11];
    const float* nw2  = (const float*)d_in[12]; const float* nb2 = (const float*)d_in[13];
    bf16_t* wsW = (bf16_t*)d_ws;               // needs 512 KB scratch
    float*  out = (float*)d_out;

    pack_weights<<<WS_ELEMS / 256, 256, 0, stream>>>(pw0, pw1, pw2, nw0, nw1, nw2, wsW);

    const int blocks = (T_DIM * N_DIM) / ROWS_PER_WAVE;   // 6250

    // pass 1: pos MLP -> rows with obs==1, plus passthrough rows (obs==0)
    mlp_pass<<<blocks, 32, 0, stream>>>(node, obs,
        wsW + OFF_PW0, wsW + OFF_PW1, wsW + OFF_PW2,
        pb0, pb1, pb2, out, +1, 1);

    // pass 2: neg MLP -> rows with obs==-1
    mlp_pass<<<blocks, 32, 0, stream>>>(node, obs,
        wsW + OFF_NW0, wsW + OFF_NW1, wsW + OFF_NW2,
        nb0, nb1, nb2, out, -1, 0);
}
